// SparseMultiResUNet_75814762709627
// MI455X (gfx1250) — matrix operand order repaired and checked by reading, not compiled, not run on hardware
//
#include <hip/hip_runtime.h>
#include <hip/hip_bf16.h>

typedef __attribute__((ext_vector_type(16))) __bf16 v16bf;
typedef __attribute__((ext_vector_type(8)))  float  v8f;

#define EPS_F 1e-5f

// A fragment k order used here (contiguous, B matches): lane holds row m=lane&15,
// element e -> K = e + 16*(lane>>4)
__device__ __forceinline__ int a_kmap(int e, int half) { return e + 16 * half; }

// ---------------------------------------------------------------------------
// Gather-conv:  out[N,Cout] = sum_o gather(x, nbr[:,o]) @ W[o]
// x: bf16 activations; W: bf16 weights; out: fp32 (WMMA f32 accumulator).
// nbr==nullptr -> identity (linear, K==1).
// 4 waves/block, each wave owns one 16-row tile and computes TWO 16-wide
// cout tiles from a single A fragment (A gather reuse); B tiles via LDS.
// ---------------------------------------------------------------------------
__global__ __launch_bounds__(128) void gconv_wmma(
    const __bf16* __restrict__ x, const int* __restrict__ nbr,
    const __bf16* __restrict__ W, float* __restrict__ out,
    int N, int Cin, int Cout, int K)
{
    __shared__ __bf16 lsb[1024];                // two B tiles: [tile][k=0..31][n=0..15]
    const int wave = threadIdx.x >> 5;
    const int lane = threadIdx.x & 31;
    const int half = lane >> 4;
    const int l16  = lane & 15;
    const int ct0  = blockIdx.y * 2;            // first cout tile
    const bool has2 = (ct0 + 1) * 16 < Cout;    // second tile exists (block-uniform)
    const int m0   = (blockIdx.x * 4 + wave) * 16;

    v8f acc0 = {}, acc1 = {};
    const int tot = has2 ? 1024 : 512;
    for (int o = 0; o < K; ++o) {
        for (int c0 = 0; c0 < Cin; c0 += 32) {
            // cooperative B tile(s) load -> LDS
            for (int t = threadIdx.x; t < tot; t += 128) {
                int tile = t >> 9, r = t & 511;
                int k = r >> 4, nn = r & 15;
                int c = c0 + k;
                __bf16 v = (__bf16)0.f;
                if (c < Cin) v = W[((size_t)o * Cin + c) * Cout + (ct0 + tile) * 16 + nn];
                lsb[t] = v;
            }
            __syncthreads();

            // per-lane A fragment: two contiguous 8xbf16 runs -> b128 loads
            v16bf a;
            int row = m0 + l16;
            int src = -1;
            if (row < N) src = nbr ? nbr[(size_t)row * K + o] : row;
            if (src >= 0) {
                const __bf16* bp = x + (size_t)src * Cin + c0;
                if (c0 + 32 <= Cin) {
#pragma unroll
                    for (int e = 0; e < 16; ++e) a[e] = bp[a_kmap(e, half)];
                } else {
#pragma unroll
                    for (int e = 0; e < 16; ++e) {
                        int k = a_kmap(e, half);
                        a[e] = (c0 + k < Cin) ? bp[k] : (__bf16)0.f;
                    }
                }
            } else {
#pragma unroll
                for (int e = 0; e < 16; ++e) a[e] = (__bf16)0.f;
            }

            // B fragments from LDS: element e -> K = e + 16*half, col = l16
            v16bf b0;
#pragma unroll
            for (int e = 0; e < 16; ++e) b0[e] = lsb[((e + 16 * half) << 4) + l16];
            acc0 = __builtin_amdgcn_wmma_f32_16x16x32_bf16(
                false, a, false, b0, (short)0, acc0, false, false);
            if (has2) {
                v16bf b1;
#pragma unroll
                for (int e = 0; e < 16; ++e) b1[e] = lsb[512 + (((e + 16 * half) << 4) + l16)];
                acc1 = __builtin_amdgcn_wmma_f32_16x16x32_bf16(
                    false, a, false, b1, (short)0, acc1, false, false);
            }
            __syncthreads();
        }
    }
    // D layout: vgpr v -> row m = v + 8*half, col = l16
#pragma unroll
    for (int v = 0; v < 8; ++v) {
        int m = m0 + v + 8 * half;
        if (m < N) {
            out[(size_t)m * Cout + ct0 * 16 + l16] = acc0[v];
            if (has2) out[(size_t)m * Cout + (ct0 + 1) * 16 + l16] = acc1[v];
        }
    }
}

// ---------------------------------------------------------------------------
// Transposed conv (scatter): out[dn[i,o]] += (x @ W[o])[i]   one wave / tile
// x, W: bf16; out: fp32 accumulator (pre-zeroed), atomicAdd scatter.
// ---------------------------------------------------------------------------
__global__ __launch_bounds__(32) void upconv_wmma(
    const __bf16* __restrict__ x, const int* __restrict__ dn,
    const __bf16* __restrict__ W, float* __restrict__ out,
    int Nc, int Cin, int Cout)
{
    const int lane = threadIdx.x;
    const int half = lane >> 4;
    const int l16  = lane & 15;
    const int ct   = blockIdx.y;
    const int o    = blockIdx.z;
    const int m0   = blockIdx.x * 16;
    const int ncol = ct * 16 + l16;

    v8f acc = {};
    for (int c0 = 0; c0 < Cin; c0 += 32) {
        v16bf a, b;
        int row = m0 + l16;
        if (row < Nc) {
            const __bf16* bp = x + (size_t)row * Cin + c0;
            if (c0 + 32 <= Cin) {
#pragma unroll
                for (int e = 0; e < 16; ++e) a[e] = bp[a_kmap(e, half)];
            } else {
#pragma unroll
                for (int e = 0; e < 16; ++e) {
                    int k = a_kmap(e, half);
                    a[e] = (c0 + k < Cin) ? bp[k] : (__bf16)0.f;
                }
            }
        } else {
#pragma unroll
            for (int e = 0; e < 16; ++e) a[e] = (__bf16)0.f;
        }
#pragma unroll
        for (int e = 0; e < 16; ++e) {
            int c = c0 + e + 16 * half;
            b[e] = (c < Cin) ? W[((size_t)o * Cin + c) * Cout + ncol] : (__bf16)0.f;
        }
        acc = __builtin_amdgcn_wmma_f32_16x16x32_bf16(
            false, a, false, b, (short)0, acc, false, false);
    }
#pragma unroll
    for (int v = 0; v < 8; ++v) {
        int m = m0 + v + 8 * half;
        if (m < Nc) {
            int dst = dn[(size_t)m * 8 + o];
            if (dst >= 0) atomicAdd(&out[(size_t)dst * Cout + ncol], acc[v]);
        }
    }
}

// ---------------------------------------------------------------------------
// BatchNorm: stats reduce (LDS partials) + fused apply / residual / relu.
// apply reads fp32 conv output, writes bf16 activation (fused down-convert).
// ---------------------------------------------------------------------------
__global__ __launch_bounds__(256) void bn_reduce(
    const float* __restrict__ x, int N, int C, float* __restrict__ stats)
{
    __shared__ float ls[384];
    for (int i = threadIdx.x; i < 2 * C; i += 256) ls[i] = 0.f;
    __syncthreads();
    int total = N * C;
    for (int i = blockIdx.x * 256 + threadIdx.x; i < total; i += gridDim.x * 256) {
        float v = x[i];
        int c = i % C;
        atomicAdd(&ls[c], v);
        atomicAdd(&ls[C + c], v * v);
    }
    __syncthreads();
    for (int i = threadIdx.x; i < 2 * C; i += 256) atomicAdd(&stats[i], ls[i]);
}

__global__ __launch_bounds__(256) void bn_apply(
    const float* __restrict__ x, const float* __restrict__ stats,
    const float* __restrict__ g, const float* __restrict__ b,
    const __bf16* __restrict__ res, __bf16* __restrict__ out,
    int N, int C, int relu)
{
    int i = blockIdx.x * 256 + threadIdx.x;
    if (i >= N * C) return;
    int c = i % C;
    float invN = 1.f / (float)N;
    float mu   = stats[c] * invN;
    float var  = stats[C + c] * invN - mu * mu;
    float v = (x[i] - mu) * rsqrtf(var + EPS_F) * g[c] + b[c];
    if (res)  v += (float)res[i];
    if (relu) v = fmaxf(v, 0.f);
    out[i] = (__bf16)v;
}

__global__ __launch_bounds__(256) void fill_zero(float* __restrict__ p, int n)
{
    int i = blockIdx.x * 256 + threadIdx.x;
    if (i < n) p[i] = 0.f;
}

__global__ __launch_bounds__(256) void cvt_f2b(
    const float* __restrict__ in, __bf16* __restrict__ out, int n)
{
    int i = blockIdx.x * 256 + threadIdx.x;
    if (i < n) out[i] = (__bf16)in[i];
}

__global__ __launch_bounds__(256) void concat2(
    const __bf16* __restrict__ a, const __bf16* __restrict__ b,
    __bf16* __restrict__ out, int N, int Ca, int Cb)
{
    int C = Ca + Cb;
    int i = blockIdx.x * 256 + threadIdx.x;
    if (i >= N * C) return;
    int r = i / C, c = i % C;
    out[i] = (c < Ca) ? a[(size_t)r * Ca + c] : b[(size_t)r * Cb + (c - Ca)];
}

__global__ __launch_bounds__(256) void head_lin(
    const __bf16* __restrict__ y, const float* __restrict__ W,
    float* __restrict__ out, int N, int C, int ncls)
{
    int i = blockIdx.x * 256 + threadIdx.x;
    if (i >= N * ncls) return;
    int r = i / ncls, c = i % ncls;
    float acc = 0.f;
    for (int k = 0; k < C; ++k)
        acc = fmaf((float)y[(size_t)r * C + k], W[(size_t)k * ncls + c], acc);
    out[i] = acc;
}

// ---------------------------------------------------------------------------
// Host orchestration
// ---------------------------------------------------------------------------
namespace {

struct BNp { const float* g; const float* b; };
struct RBp { const __bf16* c0w; BNp bn0; const __bf16* c1w; BNp bn1;
             const __bf16* scw; BNp scbn; bool has_sc; };

inline void launch_gconv(hipStream_t st, const __bf16* x, const int* nbr, const __bf16* W,
                         float* out, int N, int Cin, int Cout, int K) {
    dim3 grid((N + 63) / 64, (Cout / 16 + 1) / 2);
    gconv_wmma<<<grid, 128, 0, st>>>(x, nbr, W, out, N, Cin, Cout, K);
}

inline void launch_zero(hipStream_t st, float* p, size_t n) {
    fill_zero<<<(unsigned)((n + 255) / 256), 256, 0, st>>>(p, (int)n);
}

inline void launch_bn(hipStream_t st, const float* x, int N, int C, BNp bn,
                      const __bf16* res, int relu, __bf16* out, float* stats) {
    launch_zero(st, stats, (size_t)2 * C);
    int total = N * C;
    int rb = (total + 255) / 256; if (rb > 2048) rb = 2048;
    bn_reduce<<<rb, 256, 0, st>>>(x, N, C, stats);
    bn_apply<<<(total + 255) / 256, 256, 0, st>>>(x, stats, bn.g, bn.b, res, out, N, C, relu);
}

// out (bf16) = relu(bn1(conv1(relu(bn0(conv0(xin))))) + shortcut(xin))
inline void resblock(hipStream_t st, const __bf16* xin, int N, int Cin, int Cout,
                     const int* nbr, const RBp& rb, __bf16* out,
                     float* f0, float* f1, __bf16* btmp, float* stats) {
    launch_gconv(st, xin, nbr, rb.c0w, f0, N, Cin, Cout, 27);
    launch_bn(st, f0, N, Cout, rb.bn0, nullptr, 1, btmp, stats);
    launch_gconv(st, btmp, nbr, rb.c1w, f1, N, Cout, Cout, 27);
    const __bf16* s;
    if (rb.has_sc) {
        launch_gconv(st, xin, nullptr, rb.scw, f0, N, Cin, Cout, 1);  // linear
        launch_bn(st, f0, N, Cout, rb.scbn, nullptr, 0, out, stats);  // s -> out (temp)
        s = out;
    } else {
        s = xin;
    }
    launch_bn(st, f1, N, Cout, rb.bn1, s, 1, out, stats);
}

} // namespace

extern "C" void kernel_launch(void* const* d_in, const int* in_sizes, int n_in,
                              void* d_out, int out_size, void* d_ws, size_t ws_size,
                              hipStream_t stream) {
    (void)n_in; (void)out_size; (void)ws_size;
    static const int NC[9]   = {16, 16, 32, 64, 128, 128, 64, 48, 48};
    static const int NCLS[5] = {3, 4, 7, 8, 14};

    // ---- workspace carving helper (used for weights during the walk) ----
    char* ws = (char*)d_ws;
    size_t off = 0;
    auto walloc = [&](size_t bytes) {
        char* r = ws + off; off += (bytes + 255) & ~(size_t)255; return r;
    };

    int p = 0;
    const float* feats = (const float*)d_in[p]; int n0 = in_sizes[p] / 4; p++;

    auto nextf = [&]() { return (const float*)d_in[p++]; };
    // convert a weight tensor to a bf16 workspace copy (once per launch)
    auto cw = [&](const float* w, size_t n) -> const __bf16* {
        __bf16* d = (__bf16*)walloc(n * sizeof(__bf16));
        cvt_f2b<<<(unsigned)((n + 255) / 256), 256, 0, stream>>>(w, d, (int)n);
        return d;
    };
    auto getBN = [&]() { BNp r; r.g = nextf(); r.b = nextf(); return r; };
    auto getRB = [&](int cin, int cout) {
        RBp r;
        r.c0w = cw(nextf(), (size_t)27 * cin * cout);  r.bn0 = getBN();
        r.c1w = cw(nextf(), (size_t)27 * cout * cout); r.bn1 = getBN();
        r.has_sc = (cin != cout); r.scw = nullptr;
        if (r.has_sc) { r.scw = cw(nextf(), (size_t)cin * cout); r.scbn = getBN(); }
        return r;
    };

    const __bf16* stem_c0 = cw(nextf(), (size_t)27 * 4 * 16);  BNp stem_bn0 = getBN();
    const __bf16* stem_c1 = cw(nextf(), (size_t)27 * 16 * 16); BNp stem_bn1 = getBN();
    struct Enc { const __bf16* dw; BNp dbn; RBp rb0, rb1; } enc[4];
    for (int k = 0; k < 4; ++k) {
        enc[k].dw = cw(nextf(), (size_t)8 * NC[k] * NC[k]); enc[k].dbn = getBN();
        enc[k].rb0 = getRB(NC[k], NC[k + 1]);
        enc[k].rb1 = getRB(NC[k + 1], NC[k + 1]);
    }
    struct Dec { const __bf16* uw; BNp ubn; RBp rb0, rb1; } dec[5][4];
    const float* headw[5];
    for (int h = 0; h < 5; ++h) {
        for (int k = 0; k < 4; ++k) {
            dec[h][k].uw = cw(nextf(), (size_t)8 * NC[k + 4] * NC[k + 5]);
            dec[h][k].ubn = getBN();
            dec[h][k].rb0 = getRB(NC[k + 5] + NC[3 - k], NC[k + 5]);
            dec[h][k].rb1 = getRB(NC[k + 5], NC[k + 5]);
        }
        headw[h] = nextf();
    }
    const int* nbr3[5]; int nl[5];
    for (int l = 0; l < 5; ++l) { nbr3[l] = (const int*)d_in[p]; nl[l] = in_sizes[p] / 27; p++; }
    const int* dnb[4];
    for (int l = 0; l < 4; ++l) dnb[l] = (const int*)d_in[p++];
    if (nl[0] <= 0) nl[0] = n0;

    // ---- activation workspace ----
    size_t cap = 0;
    {
        size_t c[5] = {(size_t)nl[0] * 64, (size_t)nl[1] * 64, (size_t)nl[2] * 96,
                       (size_t)nl[3] * 192, (size_t)nl[4] * 128};
        for (int i = 0; i < 5; ++i) if (c[i] > cap) cap = c[i];
    }
    float* stats = (float*)walloc(512 * sizeof(float));
    float* F0    = (float*)walloc(cap * sizeof(float));   // conv raw / upconv accum
    float* F1    = (float*)walloc(cap * sizeof(float));
    __bf16* B0   = (__bf16*)walloc(cap * sizeof(__bf16));
    __bf16* B1   = (__bf16*)walloc(cap * sizeof(__bf16));
    __bf16* B2   = (__bf16*)walloc(cap * sizeof(__bf16));
    __bf16* featsb = (__bf16*)walloc((size_t)nl[0] * 4 * sizeof(__bf16));
    __bf16* codes[5];
    codes[0] = (__bf16*)walloc((size_t)nl[0] * 16  * sizeof(__bf16));
    codes[1] = (__bf16*)walloc((size_t)nl[1] * 16  * sizeof(__bf16));
    codes[2] = (__bf16*)walloc((size_t)nl[2] * 32  * sizeof(__bf16));
    codes[3] = (__bf16*)walloc((size_t)nl[3] * 64  * sizeof(__bf16));
    codes[4] = (__bf16*)walloc((size_t)nl[4] * 128 * sizeof(__bf16));

    // ---- stem ----
    {
        int n = nl[0] * 4;
        cvt_f2b<<<(n + 255) / 256, 256, 0, stream>>>(feats, featsb, n);
    }
    launch_gconv(stream, featsb, nbr3[0], stem_c0, F0, nl[0], 4, 16, 27);
    launch_bn(stream, F0, nl[0], 16, stem_bn0, nullptr, 1, B0, stats);
    launch_gconv(stream, B0, nbr3[0], stem_c1, F1, nl[0], 16, 16, 27);
    launch_bn(stream, F1, nl[0], 16, stem_bn1, nullptr, 1, codes[0], stats);

    // ---- encoder ----
    for (int k = 0; k < 4; ++k) {
        int Nc = nl[k + 1];
        launch_gconv(stream, codes[k], dnb[k], enc[k].dw, F0, Nc, NC[k], NC[k], 8);
        launch_bn(stream, F0, Nc, NC[k], enc[k].dbn, nullptr, 1, B0, stats);
        resblock(stream, B0, Nc, NC[k], NC[k + 1], nbr3[k + 1], enc[k].rb0,
                 B1, F0, F1, B2, stats);
        resblock(stream, B1, Nc, NC[k + 1], NC[k + 1], nbr3[k + 1], enc[k].rb1,
                 codes[k + 1], F0, F1, B2, stats);
    }

    // ---- decoders ----
    float* outf = (float*)d_out;
    size_t out_off = 0;
    for (int h = 0; h < 5; ++h) {
        const __bf16* y = codes[4];
        int Ny = nl[4], Cy = 128;
        for (int k = 0; k < 4; ++k) {
            int lvl = 3 - k;
            int Nf = nl[lvl];
            int Cu = NC[k + 5];
            int Cc = NC[lvl];
            // u = relu(bn(up_conv(y)))
            launch_zero(stream, F0, (size_t)Nf * Cu);
            {
                dim3 grid((Ny + 15) / 16, Cu / 16, 8);
                upconv_wmma<<<grid, 32, 0, stream>>>(y, dnb[lvl], dec[h][k].uw, F0, Ny, Cy, Cu);
            }
            launch_bn(stream, F0, Nf, Cu, dec[h][k].ubn, nullptr, 1, B0, stats);
            // concat [u | codes[lvl]]
            {
                int total = Nf * (Cu + Cc);
                concat2<<<(total + 255) / 256, 256, 0, stream>>>(B0, codes[lvl], B1, Nf, Cu, Cc);
            }
            resblock(stream, B1, Nf, Cu + Cc, Cu, nbr3[lvl], dec[h][k].rb0,
                     B2, F0, F1, B0, stats);
            resblock(stream, B2, Nf, Cu, Cu, nbr3[lvl], dec[h][k].rb1,
                     B1, F0, F1, B0, stats);
            y = B1; Ny = Nf; Cy = Cu;
        }
        // head: (n0 x 48) @ (48 x ncls)
        {
            int ncls = NCLS[h];
            int total = nl[0] * ncls;
            head_lin<<<(total + 255) / 256, 256, 0, stream>>>(y, headw[h], outf + out_off,
                                                              nl[0], 48, ncls);
            out_off += (size_t)nl[0] * ncls;
        }
    }
}
